// GRU_64218351010319
// MI455X (gfx1250) — matrix operand order repaired and checked by reading, not compiled, not run on hardware
//
#include <hip/hip_runtime.h>

typedef __attribute__((ext_vector_type(8)))  float  f32x8;
typedef __attribute__((ext_vector_type(8)))  __bf16 bf16x8;
typedef __attribute__((ext_vector_type(16))) __bf16 bf16x16;

#define Bsz  256
#define Ssz  1024
#define Fin  75
#define Hd   256
#define G3   768          // 3*H gate width
#define Mtok (Bsz*Ssz)    // 262144 tokens
#define Kpad 96           // F=75 padded to multiple of 32

// ---------- helpers ----------
static __device__ __forceinline__ unsigned short f32_bf16(float f) {
  unsigned u = __builtin_bit_cast(unsigned, f);
  unsigned r = u + 0x7FFFu + ((u >> 16) & 1u);   // round-to-nearest-even
  return (unsigned short)(r >> 16);
}

static __device__ __forceinline__ bf16x16 load_frag16(const unsigned short* p0,
                                                      const unsigned short* p1) {
  bf16x8 lo = *(const bf16x8*)p0;   // 16B
  bf16x8 hi = *(const bf16x8*)p1;   // 16B
  return __builtin_shufflevector(lo, hi, 0,1,2,3,4,5,6,7,8,9,10,11,12,13,14,15);
}

static __device__ __forceinline__ float sigmoidf_(float x) {
  return 1.0f / (1.0f + __expf(-x));
}

// ---------- pack kernels ----------
__global__ void k_cast_bf16(const float* __restrict__ src,
                            unsigned short* __restrict__ dst, long n) {
  long i = (long)blockIdx.x * blockDim.x + threadIdx.x;
  if (i < n) dst[i] = f32_bf16(src[i]);
}

__global__ void k_pad_cast_bf16(const float* __restrict__ src,
                                unsigned short* __restrict__ dst,
                                int src_cols, int dst_cols, long n) {
  long i = (long)blockIdx.x * blockDim.x + threadIdx.x;
  if (i >= n) return;
  long row = i / dst_cols;
  int  c   = (int)(i - row * dst_cols);
  dst[i] = (c < src_cols) ? f32_bf16(src[row * src_cols + c]) : (unsigned short)0;
}

// ---------- big WMMA GEMM: C[M,768] = A[M,K](bf16) * W[768,K]^T(bf16) + bias ----
template<int K>
__global__ void k_wmma_gemm_bias(const unsigned short* __restrict__ A,  // lda = K
                                 const unsigned short* __restrict__ W,  // ldb = K
                                 const float* __restrict__ bias,
                                 float* __restrict__ C, int ldc, int M) {
  constexpr int NKB = K / 32;
  const int lane  = threadIdx.x & 31;
  const int wave  = threadIdx.x >> 5;
  const int lidx  = lane & 15;
  const int khalf = lane >> 4;
  const long Wv   = (long)blockIdx.x * 8 + wave;
  const int  NG   = G3 / 64;                 // 12 n-groups
  const long mt   = Wv / NG;
  const int  ng   = (int)(Wv % NG);
  if (mt >= (M >> 4)) return;                // uniform per wave

  f32x8 acc[4];
  #pragma unroll
  for (int j = 0; j < 4; ++j) {
    float bv = bias[ng * 64 + j * 16 + lidx];
    #pragma unroll
    for (int v = 0; v < 8; ++v) acc[j][v] = bv;
  }

  // preload entire A panel for this wave (<= 8 frags = 64 VGPRs)
  bf16x16 af[NKB];
  const unsigned short* arow = A + ((long)mt * 16 + lidx) * K + khalf * 8;
  #pragma unroll
  for (int kb = 0; kb < NKB; ++kb)
    af[kb] = load_frag16(arow + kb * 32, arow + kb * 32 + 16);

  #pragma unroll
  for (int kb = 0; kb < NKB; ++kb) {
    bf16x16 bfv[4];
    #pragma unroll
    for (int j = 0; j < 4; ++j) {
      const unsigned short* brow =
          W + (long)(ng * 64 + j * 16 + lidx) * K + kb * 32 + khalf * 8;
      bfv[j] = load_frag16(brow, brow + 16);
    }
    #pragma unroll
    for (int j = 0; j < 4; ++j)
      acc[j] = __builtin_amdgcn_wmma_f32_16x16x32_bf16(
          false, af[kb], false, bfv[j], (short)0, acc[j], false, false);
  }

  const int mrow0 = (int)mt * 16 + 8 * khalf;
  #pragma unroll
  for (int j = 0; j < 4; ++j) {
    int n = ng * 64 + j * 16 + lidx;
    #pragma unroll
    for (int v = 0; v < 8; ++v)
      C[(long)(mrow0 + v) * ldc + n] = acc[j][v];
  }
}

// ---------- persistent GRU recurrence ----------
// 128 blocks x 64 threads = 256 waves; wave (mt,ct) owns a 16x16 h-tile and its
// three gate tiles (r,z,n).  w_hh fragments (24KB/wave, time-invariant) are
// staged once into LDS in WMMA fragment order.  Per step: A-panel loads first,
// gate-phase xp/h loads second (hidden behind the WMMA chain), ds reads of B
// software-pipelined 2 deep ahead of their consuming WMMAs.
static __device__ __forceinline__ void grid_sync(unsigned* cnt, unsigned target) {
  __threadfence();
  __syncthreads();
  if (threadIdx.x == 0) {
    atomicAdd(cnt, 1u);
    while (__hip_atomic_load(cnt, __ATOMIC_RELAXED, __HIP_MEMORY_SCOPE_AGENT) < target) {
      __builtin_amdgcn_s_sleep(1);
    }
  }
  __syncthreads();
  __threadfence();
}

__global__ void k_gru_rec(const float* __restrict__ xp,            // [Mtok, 768] f32
                          const unsigned short* __restrict__ whh,  // [768, 256] bf16
                          const float* __restrict__ bhh,           // [768]
                          float* __restrict__ hf,                  // [2][256*256] f32
                          unsigned short* __restrict__ hb,         // [2][256*256] bf16
                          unsigned short* __restrict__ hseq,       // [Mtok, 256] bf16
                          unsigned* __restrict__ sync_cnt) {
  extern __shared__ unsigned short lsB[];    // 2 waves * 24KB
  const int lane  = threadIdx.x & 31;
  const int wave  = threadIdx.x >> 5;        // 0..1
  const int Wv    = blockIdx.x * 2 + wave;   // 0..255
  const int mt    = Wv >> 4;                 // batch tile
  const int ct    = Wv & 15;                 // h-column tile
  const int lidx  = lane & 15;
  const int khalf = lane >> 4;
  const int colg  = ct * 16 + lidx;          // h column 0..255

  // ---- stage this wave's w_hh fragments into LDS (once), frag i = kb*3+r ----
  unsigned short* myB = lsB + wave * (3 * 8 * 512);   // 512 ushorts per frag
  #pragma unroll
  for (int kb = 0; kb < 8; ++kb) {
    #pragma unroll
    for (int r = 0; r < 3; ++r) {
      const unsigned short* brow =
          whh + (long)(r * Hd + colg) * Hd + kb * 32 + khalf * 8;
      bf16x8 lo = *(const bf16x8*)brow;
      bf16x8 hi = *(const bf16x8*)(brow + 16);
      unsigned short* d = myB + (kb * 3 + r) * 512 + lane * 16;
      *(bf16x8*)d       = lo;
      *(bf16x8*)(d + 8) = hi;
    }
  }
  __syncthreads();

  float bias[3];
  #pragma unroll
  for (int r = 0; r < 3; ++r) bias[r] = bhh[r * Hd + colg];

  const unsigned short* myBl = myB + lane * 16;

  for (int t = 0; t < Ssz; ++t) {
    const int cur = t & 1, nxt = cur ^ 1;
    const unsigned short* H = hb + cur * (Bsz * Hd);

    // (1) A panel loads FIRST (16 x global_load_b128)
    bf16x16 af[8];
    const unsigned short* arow = H + (mt * 16 + lidx) * Hd + khalf * 8;
    #pragma unroll
    for (int kb = 0; kb < 8; ++kb)
      af[kb] = load_frag16(arow + kb * 32, arow + kb * 32 + 16);

    // (2) gate-phase operands SECOND (consumed only after the WMMA chain,
    //     so their latency hides behind it; LOADcnt FIFO keeps A waits early)
    float xr[8], xz[8], xn[8], ho[8];
    #pragma unroll
    for (int v = 0; v < 8; ++v) {
      const int  m    = mt * 16 + 8 * khalf + v;
      const long xrow = ((long)m * Ssz + t) * G3;
      xr[v] = xp[xrow + colg];
      xz[v] = xp[xrow + Hd + colg];
      xn[v] = xp[xrow + 2 * Hd + colg];
      ho[v] = hf[cur * (Bsz * Hd) + m * Hd + colg];
    }

    f32x8 acc[3];
    #pragma unroll
    for (int r = 0; r < 3; ++r) {
      #pragma unroll
      for (int v = 0; v < 8; ++v) acc[r][v] = bias[r];
    }

    // (3) 24-WMMA chain; ds reads of B pipelined 2 fragments ahead
    bf16x8 plo[2], phi[2];
    plo[0] = *(const bf16x8*)(myBl);
    phi[0] = *(const bf16x8*)(myBl + 8);
    plo[1] = *(const bf16x8*)(myBl + 512);
    phi[1] = *(const bf16x8*)(myBl + 512 + 8);
    #pragma unroll
    for (int i = 0; i < 24; ++i) {
      bf16x16 bfv = __builtin_shufflevector(plo[i & 1], phi[i & 1],
          0,1,2,3,4,5,6,7,8,9,10,11,12,13,14,15);
      if (i + 2 < 24) {
        plo[i & 1] = *(const bf16x8*)(myBl + (i + 2) * 512);
        phi[i & 1] = *(const bf16x8*)(myBl + (i + 2) * 512 + 8);
      }
      acc[i % 3] = __builtin_amdgcn_wmma_f32_16x16x32_bf16(
          false, af[i / 3], false, bfv, (short)0, acc[i % 3], false, false);
    }

    // (4) gate math: lane element v -> (m = mt*16 + 8*khalf + v, column colg)
    #pragma unroll
    for (int v = 0; v < 8; ++v) {
      const int m  = mt * 16 + 8 * khalf + v;
      float rg = sigmoidf_(xr[v] + acc[0][v]);
      float zg = sigmoidf_(xz[v] + acc[1][v]);
      float ng = tanhf(xn[v] + rg * acc[2][v]);
      float hn = (1.0f - zg) * ng + zg * ho[v];
      hf[nxt * (Bsz * Hd) + m * Hd + colg] = hn;
      unsigned short hv = f32_bf16(hn);
      hb[nxt * (Bsz * Hd) + m * Hd + colg] = hv;
      hseq[((long)m * Ssz + t) * Hd + colg] = hv;
    }

    if (t + 1 < Ssz) {  // warm L2 for next step's x_proj slice
      __builtin_prefetch(&xp[(((long)(mt * 16 + 8 * khalf)) * Ssz + t + 1) * G3 + colg], 0, 1);
    }

    grid_sync(sync_cnt, (unsigned)(t + 1) * gridDim.x);
  }
}

// ---------- head: out[b] = dot(h_final[b,:], w_lin) + b_lin ----------
__global__ void k_head(const float* __restrict__ hfinal,
                       const float* __restrict__ wlin,
                       const float* __restrict__ blin,
                       float* __restrict__ out) {
  const int b = threadIdx.x;
  float s = 0.0f;
  for (int k = 0; k < Hd; ++k) s += hfinal[b * Hd + k] * wlin[k];
  out[b] = s + blin[0];
}

// ---------- launch ----------
extern "C" void kernel_launch(void* const* d_in, const int* in_sizes, int n_in,
                              void* d_out, int out_size, void* d_ws, size_t ws_size,
                              hipStream_t stream) {
  (void)in_sizes; (void)n_in; (void)out_size; (void)ws_size;
  const float* x     = (const float*)d_in[0];
  const float* w_ih0 = (const float*)d_in[1];
  const float* w_hh0 = (const float*)d_in[2];
  const float* b_ih0 = (const float*)d_in[3];
  const float* b_hh0 = (const float*)d_in[4];
  const float* w_ih1 = (const float*)d_in[5];
  const float* w_hh1 = (const float*)d_in[6];
  const float* b_ih1 = (const float*)d_in[7];
  const float* b_hh1 = (const float*)d_in[8];
  const float* w_lin = (const float*)d_in[9];
  const float* b_lin = (const float*)d_in[10];
  float* out = (float*)d_out;

  char* p = (char*)d_ws;
  auto take = [&](size_t bytes) -> char* {
    char* r = p; p += (bytes + 255) & ~(size_t)255; return r;
  };
  float*          xp    = (float*)take((size_t)Mtok * G3 * 4);          // 768 MB (reused L0/L1)
  unsigned short* xpad  = (unsigned short*)take((size_t)Mtok * Kpad * 2);
  unsigned short* wih0b = (unsigned short*)take((size_t)G3 * Kpad * 2);
  unsigned short* whh0b = (unsigned short*)take((size_t)G3 * Hd * 2);
  unsigned short* wih1b = (unsigned short*)take((size_t)G3 * Hd * 2);
  unsigned short* whh1b = (unsigned short*)take((size_t)G3 * Hd * 2);
  float*          hf    = (float*)take(2ull * Bsz * Hd * 4);
  unsigned short* hb    = (unsigned short*)take(2ull * Bsz * Hd * 2);
  unsigned short* hseq  = (unsigned short*)take((size_t)Mtok * Hd * 2); // 128 MB (reused)
  unsigned*       syncc = (unsigned*)take(256);

  // pack inputs/weights to bf16
  { long n = (long)Mtok * Kpad;
    k_pad_cast_bf16<<<(unsigned)((n + 255) / 256), 256, 0, stream>>>(x, xpad, Fin, Kpad, n); }
  { long n = (long)G3 * Kpad;
    k_pad_cast_bf16<<<(unsigned)((n + 255) / 256), 256, 0, stream>>>(w_ih0, wih0b, Fin, Kpad, n); }
  { long n = (long)G3 * Hd;
    k_cast_bf16<<<(unsigned)((n + 255) / 256), 256, 0, stream>>>(w_hh0, whh0b, n);
    k_cast_bf16<<<(unsigned)((n + 255) / 256), 256, 0, stream>>>(w_ih1, wih1b, n);
    k_cast_bf16<<<(unsigned)((n + 255) / 256), 256, 0, stream>>>(w_hh1, whh1b, n); }

  const unsigned gemm_blocks = (Mtok / 16) * (G3 / 64) / 8;  // 24576
  const size_t   rec_lds     = 2ull * 3 * 8 * 512 * sizeof(unsigned short); // 48 KB

  // layer 0: x-projection GEMM, then persistent recurrence
  k_wmma_gemm_bias<Kpad><<<gemm_blocks, 256, 0, stream>>>(xpad, wih0b, b_ih0,
                                                          xp, G3, Mtok);
  hipMemsetAsync(hf, 0, 2ull * Bsz * Hd * 4, stream);
  hipMemsetAsync(hb, 0, 2ull * Bsz * Hd * 2, stream);
  hipMemsetAsync(syncc, 0, 256, stream);
  k_gru_rec<<<128, 64, rec_lds, stream>>>(xp, whh0b, b_hh0, hf, hb, hseq, syncc);

  // layer 1: projection of h1 sequence (bf16, A-matrix-ready), then recurrence
  k_wmma_gemm_bias<Hd><<<gemm_blocks, 256, 0, stream>>>(hseq, wih1b, b_ih1,
                                                        xp, G3, Mtok);
  hipMemsetAsync(hf, 0, 2ull * Bsz * Hd * 4, stream);
  hipMemsetAsync(hb, 0, 2ull * Bsz * Hd * 2, stream);
  hipMemsetAsync(syncc, 0, 256, stream);
  k_gru_rec<<<128, 64, rec_lds, stream>>>(xp, whh1b, b_hh1, hf, hb, hseq, syncc);

  // final hidden state ends in buffer 0 (S=1024 even number of swaps)
  k_head<<<1, 256, 0, stream>>>(hf, w_lin, b_lin, out);
}
